// EGCL_v_91122026152067
// MI455X (gfx1250) — matrix operand order repaired and checked by reading, not compiled, not run on hardware
//
#include <hip/hip_runtime.h>
#include <hip/hip_bf16.h>
#include <stdint.h>

typedef __attribute__((ext_vector_type(16))) _Float16 v16h;
typedef __attribute__((ext_vector_type(8)))  float    v8f;
typedef __attribute__((ext_vector_type(4)))  uint32_t u32x4;
typedef uint32_t u32;

#define N_NODES 50000
#define N_EDGES 500000
#define N_TILES (N_NODES / 16)   /* 3125 */
#define EPSV 1e-8f

union HU { u32 d[8]; u32x4 q[2]; v16h v; };

// silu via v_rcp_f32 (fast reciprocal) instead of IEEE divide
static __device__ __forceinline__ float silu_f(float x) {
  return x * __builtin_amdgcn_rcpf(1.0f + __expf(-x));
}

static __device__ __forceinline__ v8f wmma16(v16h a, v16h b, v8f c) {
  // D = A(16x32 f16) * B(32x16 f16) + C(16x16 f32)
  return __builtin_amdgcn_wmma_f32_16x16x32_f16(false, a, false, b, (short)0, c, false, false);
}

static __device__ __forceinline__ v8f splat8(float x) {
  v8f r;
#pragma unroll
  for (int i = 0; i < 8; ++i) r[i] = x;
  return r;
}

// A fragment (16x32 f16) from a row-major f16 row viewed as dwords.
// ISA layout: V0..3 -> K = half*8 + 2v,2v+1 ; V4..7 -> K = 16 + half*8 + 2(v-4)..
static __device__ __forceinline__ v16h load_a_row(const u32* row, int kc, int half) {
  HU u;
  const u32* p = row + kc * 16 + half * 4;
  u.q[0] = *(const u32x4*)(p);
  u.q[1] = *(const u32x4*)(p + 8);
  return u.v;
}

static __device__ __forceinline__ v16h load_a_lds(const _Float16* base, int rowM, int kc, int half) {
  const u32* row = (const u32*)(base + rowM * 128);
  return load_a_row(row, kc, half);
}

// A fragment from an f32 row (m_i), converting to f16 on load.
static __device__ __forceinline__ v16h load_a_f32row(const float* row, int kc, int half) {
  union { _Float16 h[16]; v16h v; } u;
#pragma unroll
  for (int v2 = 0; v2 < 8; ++v2) {
    int col = kc * 32 + ((v2 < 4) ? (half * 8 + 2 * v2) : (16 + half * 8 + 2 * (v2 - 4)));
    u.h[2 * v2]     = (_Float16)row[col];
    u.h[2 * v2 + 1] = (_Float16)row[col + 1];
  }
  return u.v;
}

// B fragment (32x16 f16) from pre-swizzled weights: per lane 8 contiguous dwords.
static __device__ __forceinline__ v16h load_b_frag(const u32* pw, int t, int kc, int KC, int lane) {
  HU u;
  const u32* p = pw + ((size_t)((t * KC + kc) * 32 + lane)) * 8;
  u.q[0] = *(const u32x4*)(p);
  u.q[1] = *(const u32x4*)(p + 4);
  return u.v;
}

// ---------------- pack / convert / init ----------------

__global__ void k_pack(const float* __restrict__ W, u32* __restrict__ dst, int KC) {
  int i = blockIdx.x * blockDim.x + threadIdx.x;
  int total = 8 * KC * 32 * 8;
  if (i >= total) return;
  int v = i & 7;
  int lane = (i >> 3) & 31;
  int rest = i >> 8;
  int kc = rest % KC;
  int t = rest / KC;
  int half = lane >> 4;
  int n = t * 16 + (lane & 15);
  int krow = kc * 32 + ((v < 4) ? (half * 8 + 2 * v) : (16 + half * 8 + 2 * (v - 4)));
  union { _Float16 h[2]; u32 d; } u;
  u.h[0] = (_Float16)W[(size_t)krow * 128 + n];
  u.h[1] = (_Float16)W[(size_t)(krow + 1) * 128 + n];
  dst[i] = u.d;
}

__global__ void k_h16(const float* __restrict__ h, _Float16* __restrict__ h16) {
  int i = blockIdx.x * blockDim.x + threadIdx.x;
  if (i < N_NODES * 128) h16[i] = (_Float16)h[i];
}

__global__ void k_init(const float* __restrict__ coord, float* __restrict__ ca,
                       float* __restrict__ mi) {
  int i = blockIdx.x * blockDim.x + threadIdx.x;
  if (i < N_NODES * 128) mi[i] = 0.0f;
  if (i < N_NODES * 3) ca[i] = coord[i];
}

// ---------------- edge kernel: one wave per 32 edges (2 M-tiles) ----------------

__global__ __launch_bounds__(32) void k_edge(
    const int* __restrict__ eidx, const float* __restrict__ coord,
    const u32* __restrict__ h16,
    const u32* __restrict__ pw_e1, const float* __restrict__ b_e1,
    const float* __restrict__ we1_rad,
    const u32* __restrict__ pw_e2, const float* __restrict__ b_e2,
    const u32* __restrict__ pw_c1, const float* __restrict__ b_c1,
    const float* __restrict__ W_c2, const float* __restrict__ b_c2,
    float* __restrict__ coord_accum, float* __restrict__ m_i) {
  __shared__ __align__(16) _Float16 sx[32 * 128];
  __shared__ __align__(16) _Float16 sy[32 * 128];
  __shared__ float sw[32];
  __shared__ int sr[32];

  const int lane = threadIdx.x;
  const int m = lane & 15;
  const int half = lane >> 4;

  // own edge: lanes 0..15 -> tile0, lanes 16..31 -> tile1 (fully coalesced)
  const int e = blockIdx.x * 32 + lane;
  const int r = eidx[e];
  const int c = eidx[N_EDGES + e];
  sr[lane] = r;

  float dx = coord[3 * r + 0] - coord[3 * c + 0];
  float dy = coord[3 * r + 1] - coord[3 * c + 1];
  float dz = coord[3 * r + 2] - coord[3 * c + 2];
  float rad = dx * dx + dy * dy + dz * dz;

  // rad per accumulator row M = v + 8*half, per tile
  float radm0[8], radm1[8];
#pragma unroll
  for (int v = 0; v < 8; ++v) {
    radm0[v] = __shfl(rad, v + 8 * half, 32);
    radm1[v] = __shfl(rad, 16 + v + 8 * half, 32);
  }

  // A-row node ids for this lane's A-fragment role (row M = lane&15 of each tile)
  int r0A = __shfl(r, m, 32), c0A = __shfl(c, m, 32);
  int r1A = __shfl(r, 16 + m, 32), c1A = __shfl(c, 16 + m, 32);

  // ---- layer1: [h_r | h_c | rad] @ W_e1 + b_e1, silu ----
  v16h A0[8], A1[8];
#pragma unroll
  for (int kc = 0; kc < 4; ++kc) {
    A0[kc]     = load_a_row(h16 + (size_t)r0A * 64, kc, half);
    A0[4 + kc] = load_a_row(h16 + (size_t)c0A * 64, kc, half);
    A1[kc]     = load_a_row(h16 + (size_t)r1A * 64, kc, half);
    A1[4 + kc] = load_a_row(h16 + (size_t)c1A * 64, kc, half);
  }
  {
    v16h Bc = load_b_frag(pw_e1, 0, 0, 8, lane);  // pipelined B prefetch
#pragma unroll
    for (int t = 0; t < 8; ++t) {
      float bb = b_e1[t * 16 + m];
      v8f acc0 = splat8(bb);
      v8f acc1 = acc0;
#pragma unroll
      for (int kc = 0; kc < 8; ++kc) {
        v16h Bn = Bc;
        if (!(t == 7 && kc == 7))
          Bn = load_b_frag(pw_e1, (kc == 7) ? t + 1 : t, (kc == 7) ? 0 : kc + 1, 8, lane);
        acc0 = wmma16(A0[kc], Bc, acc0);
        acc1 = wmma16(A1[kc], Bc, acc1);
        Bc = Bn;
      }
      float wrad = we1_rad[t * 16 + m];  // rad column of W_e1: rank-1 update
#pragma unroll
      for (int v = 0; v < 8; ++v) {
        sx[(v + 8 * half) * 128 + t * 16 + m]      = (_Float16)silu_f(acc0[v] + radm0[v] * wrad);
        sx[(16 + v + 8 * half) * 128 + t * 16 + m] = (_Float16)silu_f(acc1[v] + radm1[v] * wrad);
      }
    }
  }
  __syncthreads();

  // ---- layer2: m_ij = silu(x @ W_e2 + b_e2) ----
  v16h Ax0[4], Ax1[4];
#pragma unroll
  for (int kc = 0; kc < 4; ++kc) {
    Ax0[kc] = load_a_lds(sx, m, kc, half);
    Ax1[kc] = load_a_lds(sx, 16 + m, kc, half);
  }
  {
    v16h Bc = load_b_frag(pw_e2, 0, 0, 4, lane);
#pragma unroll
    for (int t = 0; t < 8; ++t) {
      float bb = b_e2[t * 16 + m];
      v8f acc0 = splat8(bb);
      v8f acc1 = acc0;
#pragma unroll
      for (int kc = 0; kc < 4; ++kc) {
        v16h Bn = Bc;
        if (!(t == 7 && kc == 3))
          Bn = load_b_frag(pw_e2, (kc == 3) ? t + 1 : t, (kc == 3) ? 0 : kc + 1, 4, lane);
        acc0 = wmma16(Ax0[kc], Bc, acc0);
        acc1 = wmma16(Ax1[kc], Bc, acc1);
        Bc = Bn;
      }
#pragma unroll
      for (int v = 0; v < 8; ++v) {
        sy[(v + 8 * half) * 128 + t * 16 + m]      = (_Float16)silu_f(acc0[v]);
        sy[(16 + v + 8 * half) * 128 + t * 16 + m] = (_Float16)silu_f(acc1[v]);
      }
    }
  }
  __syncthreads();

  // ---- coord head: w = silu(m_ij @ W_c1 + b_c1) @ W_c2 + b_c2 ----
  v16h Ay0[4], Ay1[4];
#pragma unroll
  for (int kc = 0; kc < 4; ++kc) {
    Ay0[kc] = load_a_lds(sy, m, kc, half);
    Ay1[kc] = load_a_lds(sy, 16 + m, kc, half);
  }
  float part0[8], part1[8];
#pragma unroll
  for (int v = 0; v < 8; ++v) { part0[v] = 0.0f; part1[v] = 0.0f; }
  {
    v16h Bc = load_b_frag(pw_c1, 0, 0, 4, lane);
#pragma unroll
    for (int t = 0; t < 8; ++t) {
      float bb = b_c1[t * 16 + m];
      v8f acc0 = splat8(bb);
      v8f acc1 = acc0;
#pragma unroll
      for (int kc = 0; kc < 4; ++kc) {
        v16h Bn = Bc;
        if (!(t == 7 && kc == 3))
          Bn = load_b_frag(pw_c1, (kc == 3) ? t + 1 : t, (kc == 3) ? 0 : kc + 1, 4, lane);
        acc0 = wmma16(Ay0[kc], Bc, acc0);
        acc1 = wmma16(Ay1[kc], Bc, acc1);
        Bc = Bn;
      }
      float wc2 = W_c2[t * 16 + m];
#pragma unroll
      for (int v = 0; v < 8; ++v) {
        part0[v] += silu_f(acc0[v]) * wc2;
        part1[v] += silu_f(acc1[v]) * wc2;
      }
    }
  }
  // reduce over N within each 16-lane half (xor masks < 16 stay in-half)
#pragma unroll
  for (int v = 0; v < 8; ++v) {
#pragma unroll
    for (int mask = 1; mask < 16; mask <<= 1) {
      part0[v] += __shfl_xor(part0[v], mask, 32);
      part1[v] += __shfl_xor(part1[v], mask, 32);
    }
  }
  if (m == 0) {
#pragma unroll
    for (int v = 0; v < 8; ++v) {
      sw[v + 8 * half]      = part0[v] + b_c2[0];
      sw[16 + v + 8 * half] = part1[v] + b_c2[0];
    }
  }
  __syncthreads();

  // coord scatter: every lane handles its own edge
  {
    float w = sw[lane];
    float invn = 1.0f / (sqrtf(rad) + EPSV);
    atomicAdd(&coord_accum[3 * r + 0], dx * invn * w);
    atomicAdd(&coord_accum[3 * r + 1], dy * invn * w);
    atomicAdd(&coord_accum[3 * r + 2], dz * invn * w);
  }

  // m_i scatter: all 32x128 values, coalesced along N
  for (int i = lane; i < 32 * 128; i += 32) {
    int M = i >> 7;
    int Nn = i & 127;
    atomicAdd(&m_i[(size_t)sr[M] * 128 + Nn], (float)sy[i]);
  }
}

// ---------------- node kernel: one wave per 32 nodes (2 M-tiles) ----------------

__global__ __launch_bounds__(32) void k_node(
    const float* __restrict__ vel, const float* __restrict__ coord_accum,
    const u32* __restrict__ h16, const float* __restrict__ m_i,
    const u32* __restrict__ pw_v1, const float* __restrict__ b_v1,
    const float* __restrict__ W_v2, const float* __restrict__ b_v2,
    const u32* __restrict__ pw_n1, const float* __restrict__ b_n1,
    const u32* __restrict__ pw_n2, const float* __restrict__ b_n2,
    float* __restrict__ h_out, float* __restrict__ coord_out) {
  __shared__ __align__(16) _Float16 sx[32 * 128];
  __shared__ float sw[32];

  const int lane = threadIdx.x;
  const int m = lane & 15;
  const int half = lane >> 4;

  const int t0 = blockIdx.x * 2;
  int t1 = t0 + 1;
  if (t1 > N_TILES - 1) t1 = N_TILES - 1;  // duplicate tile: idempotent writes
  const int myTile = half ? t1 : t0;
  const int node = myTile * 16 + m;  // own node
  const int n0A = t0 * 16 + m;       // A-fragment row nodes
  const int n1A = t1 * 16 + m;

  v16h Ah0[4], Ah1[4];
#pragma unroll
  for (int kc = 0; kc < 4; ++kc) {
    Ah0[kc] = load_a_row(h16 + (size_t)n0A * 64, kc, half);
    Ah1[kc] = load_a_row(h16 + (size_t)n1A * 64, kc, half);
  }

  // ---- vel head: vw = silu(h @ W_v1 + b_v1) @ W_v2 + b_v2 ----
  float part0[8], part1[8];
#pragma unroll
  for (int v = 0; v < 8; ++v) { part0[v] = 0.0f; part1[v] = 0.0f; }
  {
    v16h Bc = load_b_frag(pw_v1, 0, 0, 4, lane);
#pragma unroll
    for (int t = 0; t < 8; ++t) {
      float bb = b_v1[t * 16 + m];
      v8f acc0 = splat8(bb);
      v8f acc1 = acc0;
#pragma unroll
      for (int kc = 0; kc < 4; ++kc) {
        v16h Bn = Bc;
        if (!(t == 7 && kc == 3))
          Bn = load_b_frag(pw_v1, (kc == 3) ? t + 1 : t, (kc == 3) ? 0 : kc + 1, 4, lane);
        acc0 = wmma16(Ah0[kc], Bc, acc0);
        acc1 = wmma16(Ah1[kc], Bc, acc1);
        Bc = Bn;
      }
      float wv2 = W_v2[t * 16 + m];
#pragma unroll
      for (int v = 0; v < 8; ++v) {
        part0[v] += silu_f(acc0[v]) * wv2;
        part1[v] += silu_f(acc1[v]) * wv2;
      }
    }
  }
#pragma unroll
  for (int v = 0; v < 8; ++v) {
#pragma unroll
    for (int mask = 1; mask < 16; mask <<= 1) {
      part0[v] += __shfl_xor(part0[v], mask, 32);
      part1[v] += __shfl_xor(part1[v], mask, 32);
    }
  }
  if (m == 0) {
#pragma unroll
    for (int v = 0; v < 8; ++v) {
      sw[v + 8 * half]      = part0[v] + b_v2[0];
      sw[16 + v + 8 * half] = part1[v] + b_v2[0];
    }
  }
  __syncthreads();
  {
    float vw = sw[lane];
#pragma unroll
    for (int d = 0; d < 3; ++d)
      coord_out[3 * node + d] = coord_accum[3 * node + d] + vel[3 * node + d] * vw;
  }

  // ---- node MLP layer1: silu([h | m_i] @ W_n1 + b_n1), K = 256 ----
  v16h Am0[4], Am1[4];
#pragma unroll
  for (int kc = 0; kc < 4; ++kc) {
    Am0[kc] = load_a_f32row(m_i + (size_t)n0A * 128, kc, half);
    Am1[kc] = load_a_f32row(m_i + (size_t)n1A * 128, kc, half);
  }
  {
    v16h Bc = load_b_frag(pw_n1, 0, 0, 8, lane);
#pragma unroll
    for (int t = 0; t < 8; ++t) {
      float bb = b_n1[t * 16 + m];
      v8f acc0 = splat8(bb);
      v8f acc1 = acc0;
#pragma unroll
      for (int kc = 0; kc < 8; ++kc) {
        v16h Bn = Bc;
        if (!(t == 7 && kc == 7))
          Bn = load_b_frag(pw_n1, (kc == 7) ? t + 1 : t, (kc == 7) ? 0 : kc + 1, 8, lane);
        v16h Af0 = (kc < 4) ? Ah0[kc] : Am0[kc - 4];
        v16h Af1 = (kc < 4) ? Ah1[kc] : Am1[kc - 4];
        acc0 = wmma16(Af0, Bc, acc0);
        acc1 = wmma16(Af1, Bc, acc1);
        Bc = Bn;
      }
#pragma unroll
      for (int v = 0; v < 8; ++v) {
        sx[(v + 8 * half) * 128 + t * 16 + m]      = (_Float16)silu_f(acc0[v]);
        sx[(16 + v + 8 * half) * 128 + t * 16 + m] = (_Float16)silu_f(acc1[v]);
      }
    }
  }
  __syncthreads();

  // ---- node MLP layer2 (no activation): h_out = x @ W_n2 + b_n2 ----
  v16h Ax0[4], Ax1[4];
#pragma unroll
  for (int kc = 0; kc < 4; ++kc) {
    Ax0[kc] = load_a_lds(sx, m, kc, half);
    Ax1[kc] = load_a_lds(sx, 16 + m, kc, half);
  }
  {
    v16h Bc = load_b_frag(pw_n2, 0, 0, 4, lane);
#pragma unroll
    for (int t = 0; t < 8; ++t) {
      float bb = b_n2[t * 16 + m];
      v8f acc0 = splat8(bb);
      v8f acc1 = acc0;
#pragma unroll
      for (int kc = 0; kc < 4; ++kc) {
        v16h Bn = Bc;
        if (!(t == 7 && kc == 3))
          Bn = load_b_frag(pw_n2, (kc == 3) ? t + 1 : t, (kc == 3) ? 0 : kc + 1, 4, lane);
        acc0 = wmma16(Ax0[kc], Bc, acc0);
        acc1 = wmma16(Ax1[kc], Bc, acc1);
        Bc = Bn;
      }
#pragma unroll
      for (int v = 0; v < 8; ++v) {
        h_out[(size_t)(t0 * 16 + v + 8 * half) * 128 + t * 16 + m] = acc0[v];
        h_out[(size_t)(t1 * 16 + v + 8 * half) * 128 + t * 16 + m] = acc1[v];
      }
    }
  }
}

// ---------------- launcher ----------------

extern "C" void kernel_launch(void* const* d_in, const int* in_sizes, int n_in,
                              void* d_out, int out_size, void* d_ws, size_t ws_size,
                              hipStream_t stream) {
  (void)in_sizes; (void)n_in; (void)out_size; (void)ws_size;
  const float* h     = (const float*)d_in[0];
  const float* coord = (const float*)d_in[1];
  const float* vel   = (const float*)d_in[2];
  const int*   eidx  = (const int*)d_in[3];
  const float* W_e1  = (const float*)d_in[4];
  const float* b_e1  = (const float*)d_in[5];
  const float* W_e2  = (const float*)d_in[6];
  const float* b_e2  = (const float*)d_in[7];
  const float* W_c1  = (const float*)d_in[8];
  const float* b_c1  = (const float*)d_in[9];
  const float* W_c2  = (const float*)d_in[10];
  const float* b_c2  = (const float*)d_in[11];
  const float* W_v1  = (const float*)d_in[12];
  const float* b_v1  = (const float*)d_in[13];
  const float* W_v2  = (const float*)d_in[14];
  const float* b_v2  = (const float*)d_in[15];
  const float* W_n1  = (const float*)d_in[16];
  const float* b_n1  = (const float*)d_in[17];
  const float* W_n2  = (const float*)d_in[18];
  const float* b_n2  = (const float*)d_in[19];

  char* ws = (char*)d_ws;
  u32* pw_e1 = (u32*)(ws + 0);        // KC=8: 64 KB
  u32* pw_e2 = (u32*)(ws + 65536);    // KC=4: 32 KB
  u32* pw_c1 = (u32*)(ws + 98304);
  u32* pw_v1 = (u32*)(ws + 131072);
  u32* pw_n1 = (u32*)(ws + 163840);   // KC=8: 64 KB
  u32* pw_n2 = (u32*)(ws + 229376);
  _Float16* h16 = (_Float16*)(ws + 262144);
  float* m_i = (float*)(ws + 262144 + (size_t)N_NODES * 128 * 2);
  float* coord_accum =
      (float*)(ws + 262144 + (size_t)N_NODES * 128 * 2 + (size_t)N_NODES * 128 * 4);

  float* h_out = (float*)d_out;
  float* coord_out = (float*)d_out + (size_t)N_NODES * 128;

  // pack weights into B-operand layout
  k_pack<<<(8 * 8 * 32 * 8 + 255) / 256, 256, 0, stream>>>(W_e1, pw_e1, 8); // rows 0..255
  k_pack<<<(8 * 4 * 32 * 8 + 255) / 256, 256, 0, stream>>>(W_e2, pw_e2, 4);
  k_pack<<<(8 * 4 * 32 * 8 + 255) / 256, 256, 0, stream>>>(W_c1, pw_c1, 4);
  k_pack<<<(8 * 4 * 32 * 8 + 255) / 256, 256, 0, stream>>>(W_v1, pw_v1, 4);
  k_pack<<<(8 * 8 * 32 * 8 + 255) / 256, 256, 0, stream>>>(W_n1, pw_n1, 8);
  k_pack<<<(8 * 4 * 32 * 8 + 255) / 256, 256, 0, stream>>>(W_n2, pw_n2, 4);

  k_h16<<<(N_NODES * 128 + 255) / 256, 256, 0, stream>>>(h, h16);
  k_init<<<(N_NODES * 128 + 255) / 256, 256, 0, stream>>>(coord, coord_accum, m_i);

  k_edge<<<N_EDGES / 32, 32, 0, stream>>>(
      eidx, coord, (const u32*)h16, pw_e1, b_e1, W_e1 + (size_t)256 * 128, pw_e2,
      b_e2, pw_c1, b_c1, W_c2, b_c2, coord_accum, m_i);

  k_node<<<(N_TILES + 1) / 2, 32, 0, stream>>>(
      vel, coord_accum, (const u32*)h16, m_i, pw_v1, b_v1, W_v2, b_v2, pw_n1,
      b_n1, pw_n2, b_n2, h_out, coord_out);
}
